// LSTM_Critic_21509196218926
// MI455X (gfx1250) — compile-verified, matrix operand order checked by reading, not run on hardware
//
#include <hip/hip_runtime.h>
#include <hip/hip_bf16.h>

// Problem constants (reference: T=512, B=256, D_IN=64, H=128)
#define TT   512
#define BB   256
#define DIN  64
#define HH   128
#define G4H  512              // 4*H
#define TB   (TT * BB)        // 131072 rows
#define NCT  32               // 4H / 16 column tiles

typedef __attribute__((ext_vector_type(16))) __bf16      v16bf;
typedef __attribute__((ext_vector_type(8)))  float       v8f;
typedef __attribute__((ext_vector_type(4)))  unsigned    uv4;

union FragU { v16bf v; struct { uv4 a, b; } q; };
union AccU  { v8f   v; struct { uv4 a, b; } q; };

// Load one 16-bit WMMA fragment (A or B) from a row-major bf16 matrix.
// Lane n<16 holds row n with K = {0..7, 16..23}; lane n+16 holds row n with
// K = {8..15, 24..31}  ->  two 16-byte loads at koff+hi*16 and +32.
static __device__ __forceinline__ v16bf load_frag(const void* base, int rowBytes,
                                                  int row, int koffBytes, int hi) {
  const char* p = (const char*)base + (size_t)row * rowBytes + koffBytes + hi * 16;
  FragU u;
  u.q.a = *(const uv4*)(p);
  u.q.b = *(const uv4*)(p + 32);
  return u.v;
}

static __device__ __forceinline__ unsigned short f2bf(float f) {
  unsigned u = __float_as_uint(f);
  u += 0x7FFFu + ((u >> 16) & 1u);       // round-to-nearest-even
  return (unsigned short)(u >> 16);
}
static __device__ __forceinline__ float bf2f(unsigned short s) {
  return __uint_as_float(((unsigned)s) << 16);
}
// Fast sigmoid/tanh: v_exp_f32 + v_rcp_f32 only (no IEEE divide sequence).
static __device__ __forceinline__ float sigm_f(float x) {
  return __builtin_amdgcn_rcpf(1.0f + __expf(-x));
}
static __device__ __forceinline__ float tanh_f(float x) {
  // 2*sigm(2x)-1; exp(-2x)->inf for very negative x, rcp(inf)=0 -> -1 (safe)
  return __builtin_fmaf(2.0f, sigm_f(2.0f * x), -1.0f);
}

// ---------------------------------------------------------------- converts
__global__ void k_f32_to_bf16(const float* __restrict__ src,
                              unsigned short* __restrict__ dst, int n) {
  int i = blockIdx.x * blockDim.x + threadIdx.x;
  if (i < n) dst[i] = f2bf(src[i]);
}
__global__ void k_bias_sum(const float* __restrict__ a, const float* __restrict__ b,
                           float* __restrict__ o, int n) {
  int i = blockIdx.x * blockDim.x + threadIdx.x;
  if (i < n) o[i] = a[i] + b[i];
}

// G scratch lives in WMMA D-fragment layout: [mTile][colTile 0..31][lane][8 f32]
static __device__ __forceinline__ float* g_tile_ptr(float* G, int mTile, int colTile, int lane) {
  return G + ((((size_t)mTile * NCT + colTile) * 32) + lane) * 8;
}

// ------------------------------------------------- parallel gate pre-GEMM
// G[M,512] = A[M,K](bf16) @ W[512,K]^T(bf16) + bias[512], output in fragment layout.
// Block = 256 threads = 8 waves; each block does one 16-row M tile,
// wave w covers column tiles {w, w+8, w+16, w+24}.
template <int K>
__global__ __launch_bounds__(256) void k_gemm_gates(
    const unsigned short* __restrict__ A, const unsigned short* __restrict__ W,
    const float* __restrict__ bias, float* __restrict__ G) {
  constexpr int KS = K / 32;
  const int mTile = blockIdx.x;
  const int w    = threadIdx.x >> 5;
  const int lane = threadIdx.x & 31;
  const int lr   = lane & 15;
  const int hi   = lane >> 4;

  v16bf Af[KS];
#pragma unroll
  for (int ks = 0; ks < KS; ++ks)
    Af[ks] = load_frag(A, K * 2, mTile * 16 + lr, ks * 64, hi);

#pragma unroll
  for (int q = 0; q < 4; ++q) {
    const int colTile = w + q * 8;
    const float bv = bias[colTile * 16 + lr];
    AccU acc;
#pragma unroll
    for (int e = 0; e < 8; ++e) acc.v[e] = bv;
#pragma unroll
    for (int ks = 0; ks < KS; ++ks) {
      v16bf Bf = load_frag(W, K * 2, colTile * 16 + lr, ks * 64, hi);
      acc.v = __builtin_amdgcn_wmma_f32_16x16x32_bf16(false, Af[ks], false, Bf,
                                                      (short)0, acc.v, false, false);
    }
    // Two contiguous non-temporal b128 stores (G is streaming scratch > L2)
    uv4* gp = (uv4*)g_tile_ptr(G, mTile, colTile, lane);
    __builtin_nontemporal_store(acc.q.a, gp);
    __builtin_nontemporal_store(acc.q.b, gp + 1);
  }
}

// --------------------------------------------------- sequential LSTM scan
// One persistent block per 16-row batch slice (recurrence is batch-independent).
// Recurrent weights in VGPR B-fragments for all 512 steps; h double-buffered in
// LDS (one barrier/step); c in registers. G seeds the WMMA accumulator.
__global__ __launch_bounds__(256) void k_lstm_scan(
    const float* __restrict__ G,              // fragment layout [TB/16][32][32][8]
    const unsigned short* __restrict__ Whh,   // [512, 128] bf16 row-major
    unsigned short* __restrict__ Hseq) {      // [T*B, 128] bf16 out, row-major
  __shared__ unsigned short hbuf[2][16 * HH]; // 2 x 4 KB ping-pong
  const int b0   = blockIdx.x * 16;
  const int w    = threadIdx.x >> 5;          // wave 0..7 -> H column tile
  const int lane = threadIdx.x & 31;
  const int lr   = lane & 15;
  const int hi   = lane >> 4;

  for (int i = threadIdx.x; i < 16 * HH; i += blockDim.x) hbuf[0][i] = 0;

  // Preload W_hh fragments: gate g, k-step ks -> rows g*128 + w*16 + lane
  v16bf Bf[4][4];
#pragma unroll
  for (int g = 0; g < 4; ++g)
#pragma unroll
    for (int ks = 0; ks < 4; ++ks)
      Bf[g][ks] = load_frag(Whh, HH * 2, g * HH + w * 16 + lr, ks * 64, hi);

  v8f c;
#pragma unroll
  for (int e = 0; e < 8; ++e) c[e] = 0.0f;

  for (int t = 0; t < TT; ++t) {
    const int rb = t & 1;                     // read h_{t-1}
    const int wb = rb ^ 1;                    // write h_t (buffer read at t-1)
    __syncthreads();                          // h_{t-1} stores visible; frees wb
    v16bf Af[4];
#pragma unroll
    for (int ks = 0; ks < 4; ++ks)
      Af[ks] = load_frag(hbuf[rb], HH * 2, lr, ks * 64, hi);

    const int mTile = t * (BB / 16) + blockIdx.x;   // G row-tile for this step
    v8f acc[4];
#pragma unroll
    for (int g = 0; g < 4; ++g) {
      const uv4* gp = (const uv4*)g_tile_ptr((float*)G, mTile, g * 8 + w, lane);
      AccU a;
      a.q.a = __builtin_nontemporal_load(gp);
      a.q.b = __builtin_nontemporal_load(gp + 1);
      acc[g] = a.v;
#pragma unroll
      for (int ks = 0; ks < 4; ++ks)
        acc[g] = __builtin_amdgcn_wmma_f32_16x16x32_bf16(false, Af[ks], false,
                                                         Bf[g][ks], (short)0,
                                                         acc[g], false, false);
    }

    unsigned short* hout = Hseq + (size_t)(t * BB + b0 + hi * 8) * HH + w * 16 + lr;
#pragma unroll
    for (int e = 0; e < 8; ++e) {
      float iv = sigm_f(acc[0][e]);
      float fv = sigm_f(acc[1][e]);
      float gv = tanh_f(acc[2][e]);
      float ov = sigm_f(acc[3][e]);
      float cn = fv * c[e] + iv * gv;
      c[e] = cn;
      unsigned short hb = f2bf(ov * tanh_f(cn));
      hbuf[wb][(e + hi * 8) * HH + w * 16 + lr] = hb;  // D layout -> row-major h
      hout[(size_t)e * HH] = hb;
    }
  }
}

// ------------------------------------------------------------------ head
__global__ __launch_bounds__(256) void k_head(const unsigned short* __restrict__ H1,
                                              const float* __restrict__ Wout,
                                              const float* __restrict__ bout,
                                              float* __restrict__ out) {
  int idx = blockIdx.x * blockDim.x + threadIdx.x;
  if (idx >= TB) return;
  const unsigned short* h = H1 + (size_t)idx * HH;
  float acc = 0.f;
#pragma unroll
  for (int i = 0; i < HH; ++i) acc += bf2f(h[i]) * Wout[i];
  out[idx] = acc + bout[0];
}

// ---------------------------------------------------------------- launch
extern "C" void kernel_launch(void* const* d_in, const int* in_sizes, int n_in,
                              void* d_out, int out_size, void* d_ws, size_t ws_size,
                              hipStream_t stream) {
  (void)in_sizes; (void)n_in; (void)out_size; (void)ws_size;
  const float* state = (const float*)d_in[0];
  const float* W_ih0 = (const float*)d_in[1];
  const float* W_hh0 = (const float*)d_in[2];
  const float* b_ih0 = (const float*)d_in[3];
  const float* b_hh0 = (const float*)d_in[4];
  const float* W_ih1 = (const float*)d_in[5];
  const float* W_hh1 = (const float*)d_in[6];
  const float* b_ih1 = (const float*)d_in[7];
  const float* b_hh1 = (const float*)d_in[8];
  const float* W_out = (const float*)d_in[9];
  const float* b_out = (const float*)d_in[10];
  float* out = (float*)d_out;

  char* ws = (char*)d_ws;
  size_t off = 0;
  float*          G   = (float*)(ws + off);          off += (size_t)TB * G4H * 4;  // 256 MB, reused L0->L1
  unsigned short* H0  = (unsigned short*)(ws + off); off += (size_t)TB * HH * 2;
  unsigned short* H1  = (unsigned short*)(ws + off); off += (size_t)TB * HH * 2;
  unsigned short* Xb  = (unsigned short*)(ws + off); off += (size_t)TB * DIN * 2;
  unsigned short* Wi0 = (unsigned short*)(ws + off); off += (size_t)G4H * DIN * 2;
  unsigned short* Wh0 = (unsigned short*)(ws + off); off += (size_t)G4H * HH * 2;
  unsigned short* Wi1 = (unsigned short*)(ws + off); off += (size_t)G4H * HH * 2;
  unsigned short* Wh1 = (unsigned short*)(ws + off); off += (size_t)G4H * HH * 2;
  float*          bs0 = (float*)(ws + off);          off += (size_t)G4H * 4;
  float*          bs1 = (float*)(ws + off);          off += (size_t)G4H * 4;

  // fp32 -> bf16 staging
  k_f32_to_bf16<<<(TB * DIN + 255) / 256, 256, 0, stream>>>(state, Xb, TB * DIN);
  k_f32_to_bf16<<<(G4H * DIN + 255) / 256, 256, 0, stream>>>(W_ih0, Wi0, G4H * DIN);
  k_f32_to_bf16<<<(G4H * HH + 255) / 256, 256, 0, stream>>>(W_hh0, Wh0, G4H * HH);
  k_f32_to_bf16<<<(G4H * HH + 255) / 256, 256, 0, stream>>>(W_ih1, Wi1, G4H * HH);
  k_f32_to_bf16<<<(G4H * HH + 255) / 256, 256, 0, stream>>>(W_hh1, Wh1, G4H * HH);
  k_bias_sum<<<2, 256, 0, stream>>>(b_ih0, b_hh0, bs0, G4H);
  k_bias_sum<<<2, 256, 0, stream>>>(b_ih1, b_hh1, bs1, G4H);

  // Layer 0: parallel input-gate GEMM, then batch-parallel scan
  k_gemm_gates<DIN><<<TB / 16, 256, 0, stream>>>(Xb, Wi0, bs0, G);
  k_lstm_scan<<<BB / 16, 256, 0, stream>>>(G, Wh0, H0);
  // Layer 1: parallel input-gate GEMM over h0 sequence, then scan
  k_gemm_gates<HH><<<TB / 16, 256, 0, stream>>>(H0, Wi1, bs1, G);
  k_lstm_scan<<<BB / 16, 256, 0, stream>>>(G, Wh1, H1);
  // Output head
  k_head<<<(TB + 255) / 256, 256, 0, stream>>>(H1, W_out, b_out, out);
}
